// build_unet_14946486190630
// MI455X (gfx1250) — matrix operand order repaired and checked by reading, not compiled, not run on hardware
//
#include <hip/hip_runtime.h>
#include <hip/hip_bf16.h>

// ---------------------------------------------------------------------------
// 1D U-Net forward for MI455X (gfx1250).
// Convs = implicit GEMM on v_wmma_f32_16x16x32_f16:
//  - block = 16 Cout x 256 n, 4 wave32; each wave owns 4 output tiles
//    -> 4 WMMA issues per K-chunk with a shared A fragment.
//  - B fill uses uniform row pointers (scalar addresses + one shared VGPR
//    column offset -> saddr-form loads), interior blocks skip all clamping.
//  - fragments via ds_load_b128; BN+bias+ReLU folded into the epilogue.
// ---------------------------------------------------------------------------

typedef __attribute__((ext_vector_type(16))) _Float16 v16h;
typedef __attribute__((ext_vector_type(8)))  _Float16 v8h;
typedef __attribute__((ext_vector_type(4)))  _Float16 v4h;
typedef __attribute__((ext_vector_type(8)))  float    v8f;

#define EPS_BN 1e-5f
#define ASTR   40            // LDS row stride (halves) for A tile (16 x 32)
#define BSTR   40            // LDS row stride (halves) for B tile (256 x 32)

__global__ __launch_bounds__(128)
void conv_bn_relu_wmma(const float* __restrict__ x,
                       const float* __restrict__ w,
                       const float* __restrict__ bias,
                       const float* __restrict__ gamma,
                       const float* __restrict__ beta,
                       const float* __restrict__ mean,
                       const float* __restrict__ var,
                       float* __restrict__ y,
                       int Cin, int Cout, int N)
{
    __shared__ _Float16 lds_a[16 * ASTR];    // weights: [co][k]
    __shared__ _Float16 lds_b[256 * BSTR];   // im2col:  [nn][k]  (transposed)
    __shared__ float    lds_sc[16];
    __shared__ float    lds_sh[16];

    const int tid  = threadIdx.x;
    const int lane = tid & 31;
    const int wave = tid >> 5;
    const int b    = blockIdx.z;
    const int co0  = blockIdx.y * 16;
    const int nblk = blockIdx.x * 256;
    const int K    = Cin * 3;

    const float* __restrict__ xb = x + (size_t)b * Cin * N;

    // per-channel BN scale/shift (bias folded), once per block
    if (tid < 16) {
        int   co = co0 + tid;
        float sc = gamma[co] * rsqrtf(var[co] + EPS_BN);
        lds_sc[tid] = sc;
        lds_sh[tid] = beta[co] - mean[co] * sc + bias[co] * sc;
    }

    // fill-role constants
    const int a_m  = tid >> 3;          // A: output-channel row 0..15
    const int a_kq = (tid & 7) * 4;     // A: k quad within chunk
    const int nv0  = nblk + tid;        // B: column 0 (thread fills 2 columns)
    const int nv1  = nv0 + 128;

    const bool interior = (nblk > 0) && (nblk + 256 < N);

    v8f acc0 = {}, acc1 = {}, acc2 = {}, acc3 = {};

    for (int k0 = 0; k0 < K; k0 += 32) {
        const bool full = (k0 + 32 <= K);

        // ---- A tile fill: one float4 (or clamped quad) per thread ----
        {
            const size_t wrow = (size_t)(co0 + a_m) * K;
            v4h ah;
            if (full) {                               // vector load + pack cvt
                float4 wv = *(const float4*)&w[wrow + k0 + a_kq];
                ah[0] = (_Float16)wv.x; ah[1] = (_Float16)wv.y;
                ah[2] = (_Float16)wv.z; ah[3] = (_Float16)wv.w;
            } else {                                  // tail: clamp + zero-select
                float f[4];
                #pragma unroll
                for (int u = 0; u < 4; ++u) {
                    int k  = k0 + a_kq + u;
                    int kc = k < K ? k : K - 1;
                    f[u] = w[wrow + kc];
                }
                #pragma unroll
                for (int u = 0; u < 4; ++u) {
                    int k = k0 + a_kq + u;
                    ah[u] = (_Float16)(k < K ? f[u] : 0.0f);
                }
            }
            *(v4h*)&lds_a[a_m * ASTR + a_kq] = ah;    // ds_store_b64
        }

        // ---- B tile fill: thread owns 2 columns, iterates k in 8-batches ----
        if (full && interior) {
            // fast path: uniform row pointers, shared per-lane column offset
            #pragma unroll
            for (int j = 0; j < 32; j += 8) {
                float f0[8], f1[8];
                #pragma unroll
                for (int u = 0; u < 8; ++u) {
                    int k  = k0 + j + u;                 // uniform
                    int ci = k / 3;                      // uniform
                    int t  = k - ci * 3;                 // uniform
                    const float* __restrict__ rowt =
                        xb + (size_t)ci * N + (t - 1);   // uniform pointer
                    f0[u] = rowt[nv0];                   // saddr-form loads
                    f1[u] = rowt[nv1];
                }
                v8h h0, h1;
                #pragma unroll
                for (int u = 0; u < 8; ++u) {
                    h0[u] = (_Float16)f0[u];
                    h1[u] = (_Float16)f1[u];
                }
                *(v8h*)&lds_b[(size_t)tid * BSTR + j]         = h0;
                *(v8h*)&lds_b[(size_t)(tid + 128) * BSTR + j] = h1;
            }
        } else {
            // edge / tail path: clamped addresses + zero-select
            #pragma unroll
            for (int j = 0; j < 32; j += 8) {
                float f0[8], f1[8];
                int   ok0[8], ok1[8];
                #pragma unroll
                for (int u = 0; u < 8; ++u) {
                    int k  = k0 + j + u;
                    int kc = k < K ? k : K - 1;
                    int ci = kc / 3;
                    int t  = kc - ci * 3;
                    const float* __restrict__ row = xb + (size_t)ci * N;
                    int ni0 = nv0 + t - 1;
                    int ni1 = nv1 + t - 1;
                    int nx0 = ni0 < 0 ? 0 : (ni0 >= N ? N - 1 : ni0);
                    int nx1 = ni1 < 0 ? 0 : (ni1 >= N ? N - 1 : ni1);
                    f0[u]  = row[nx0];
                    f1[u]  = row[nx1];
                    ok0[u] = (k < K) & (ni0 == nx0);
                    ok1[u] = (k < K) & (ni1 == nx1);
                }
                v8h h0, h1;
                #pragma unroll
                for (int u = 0; u < 8; ++u) {
                    h0[u] = (_Float16)(ok0[u] ? f0[u] : 0.0f);
                    h1[u] = (_Float16)(ok1[u] ? f1[u] : 0.0f);
                }
                *(v8h*)&lds_b[(size_t)tid * BSTR + j]         = h0;
                *(v8h*)&lds_b[(size_t)(tid + 128) * BSTR + j] = h1;
            }
        }

        if (k0 + 32 < K)                  // global_prefetch_b8 for next chunk
            __builtin_prefetch(&xb[(size_t)((k0 + 32) / 3) * N + nblk + tid], 0, 1);
        __syncthreads();

        // ---- fragments from LDS (ds_load_b128) ----
        const int m  = lane & 15;
        const int hi = lane >> 4;

        v16h af;
        {
            v8h lo = *(const v8h*)&lds_a[m * ASTR + hi * 16];
            v8h hh = *(const v8h*)&lds_a[m * ASTR + hi * 16 + 8];
            #pragma unroll
            for (int h = 0; h < 8; ++h) { af[h] = lo[h]; af[h + 8] = hh[h]; }
        }

        const int nnb = wave * 64 + (lane & 15);     // wave's tile-0 column
        v16h bf0, bf1, bf2, bf3;
        {
            v8h p0 = *(const v8h*)&lds_b[(size_t)nnb * BSTR + hi * 16];
            v8h p1 = *(const v8h*)&lds_b[(size_t)nnb * BSTR + hi * 16 + 8];
            #pragma unroll
            for (int h = 0; h < 8; ++h) { bf0[h] = p0[h]; bf0[h + 8] = p1[h]; }
        }
        {
            v8h p0 = *(const v8h*)&lds_b[(size_t)(nnb + 16) * BSTR + hi * 16];
            v8h p1 = *(const v8h*)&lds_b[(size_t)(nnb + 16) * BSTR + hi * 16 + 8];
            #pragma unroll
            for (int h = 0; h < 8; ++h) { bf1[h] = p0[h]; bf1[h + 8] = p1[h]; }
        }
        {
            v8h p0 = *(const v8h*)&lds_b[(size_t)(nnb + 32) * BSTR + hi * 16];
            v8h p1 = *(const v8h*)&lds_b[(size_t)(nnb + 32) * BSTR + hi * 16 + 8];
            #pragma unroll
            for (int h = 0; h < 8; ++h) { bf2[h] = p0[h]; bf2[h + 8] = p1[h]; }
        }
        {
            v8h p0 = *(const v8h*)&lds_b[(size_t)(nnb + 48) * BSTR + hi * 16];
            v8h p1 = *(const v8h*)&lds_b[(size_t)(nnb + 48) * BSTR + hi * 16 + 8];
            #pragma unroll
            for (int h = 0; h < 8; ++h) { bf3[h] = p0[h]; bf3[h + 8] = p1[h]; }
        }

        acc0 = __builtin_amdgcn_wmma_f32_16x16x32_f16(
                   false, af, false, bf0, (short)0, acc0, false, false);
        acc1 = __builtin_amdgcn_wmma_f32_16x16x32_f16(
                   false, af, false, bf1, (short)0, acc1, false, false);
        acc2 = __builtin_amdgcn_wmma_f32_16x16x32_f16(
                   false, af, false, bf2, (short)0, acc2, false, false);
        acc3 = __builtin_amdgcn_wmma_f32_16x16x32_f16(
                   false, af, false, bf3, (short)0, acc3, false, false);
        __syncthreads();
    }

    // ---- epilogue: uniform base + 32-bit per-lane offset (saddr stores) ----
    {
        const int col = lane & 15;
        const int hi  = lane >> 4;
        const int na  = nblk + wave * 64 + col;
        float* __restrict__ yb2 = y + ((size_t)b * Cout + co0) * N;
        #pragma unroll
        for (int v = 0; v < 8; ++v) {
            int   row = hi * 8 + v;
            float sc  = lds_sc[row];
            float sh  = lds_sh[row];
            int   off = row * N + na;
            yb2[off]      = fmaxf(acc0[v] * sc + sh, 0.0f);
            yb2[off + 16] = fmaxf(acc1[v] * sc + sh, 0.0f);
            yb2[off + 32] = fmaxf(acc2[v] * sc + sh, 0.0f);
            yb2[off + 48] = fmaxf(acc3[v] * sc + sh, 0.0f);
        }
    }
}

// ---------------------------------------------------------------------------
// SSA: w(n) = softmax_n( max_c x[b,c,n] ); y = w * x. One block per batch row.
// ---------------------------------------------------------------------------
__global__ __launch_bounds__(256)
void ssa_kernel(const float* __restrict__ x, float* __restrict__ y, int C, int N)
{
    __shared__ float red[256];
    const int b   = blockIdx.x;
    const int tid = threadIdx.x;
    const float* xb = x + (size_t)b * C * N;
    float*       yb = y + (size_t)b * C * N;

    float lmax = -3.402823e38f;
    for (int n = tid; n < N; n += 256) {
        float m = xb[n];
        for (int c = 1; c < C; ++c) m = fmaxf(m, xb[(size_t)c * N + n]);
        lmax = fmaxf(lmax, m);
    }
    red[tid] = lmax; __syncthreads();
    for (int s = 128; s > 0; s >>= 1) {
        if (tid < s) red[tid] = fmaxf(red[tid], red[tid + s]);
        __syncthreads();
    }
    const float M = red[0]; __syncthreads();

    float lsum = 0.0f;
    for (int n = tid; n < N; n += 256) {
        float m = xb[n];
        for (int c = 1; c < C; ++c) m = fmaxf(m, xb[(size_t)c * N + n]);
        lsum += __expf(m - M);
    }
    red[tid] = lsum; __syncthreads();
    for (int s = 128; s > 0; s >>= 1) {
        if (tid < s) red[tid] += red[tid + s];
        __syncthreads();
    }
    const float invS = 1.0f / red[0]; __syncthreads();

    for (int n = tid; n < N; n += 256) {
        float m = xb[n];
        for (int c = 1; c < C; ++c) m = fmaxf(m, xb[(size_t)c * N + n]);
        float wgt = __expf(m - M) * invS;
        for (int c = 0; c < C; ++c)
            yb[(size_t)c * N + n] = xb[(size_t)c * N + n] * wgt;
    }
}

// ---------------------------------------------------------------------------
// MaxPool1d(2)
// ---------------------------------------------------------------------------
__global__ __launch_bounds__(256)
void pool_kernel(const float* __restrict__ x, float* __restrict__ y,
                 int Nout, long long total)
{
    long long i = (long long)blockIdx.x * 256 + threadIdx.x;
    if (i >= total) return;
    long long n  = i % Nout;
    long long bc = i / Nout;
    const float* p = x + bc * (size_t)(2 * Nout) + 2 * n;
    y[i] = fmaxf(p[0], p[1]);
}

// ---------------------------------------------------------------------------
// ConvTranspose1d(k=2, s=2) into channels [0,Cout) of concat buffer.
// ---------------------------------------------------------------------------
__global__ __launch_bounds__(256)
void upconv_kernel(const float* __restrict__ x, const float* __restrict__ w,
                   const float* __restrict__ bias, float* __restrict__ cat,
                   int Cin, int Cout, int Nin, int CatC, long long total)
{
    long long i = (long long)blockIdx.x * 256 + threadIdx.x;
    if (i >= total) return;                    // total = B*Cout*2*Nin
    const int Nout = 2 * Nin;
    int t  = (int)(i % Nout);
    long long r = i / Nout;
    int o  = (int)(r % Cout);
    int b  = (int)(r / Cout);
    int n  = t >> 1;
    int kk = t & 1;
    float s = bias[o];
    const float* xb = x + (size_t)b * Cin * Nin + n;
    const float* wb = w + (size_t)o * 2 + kk;
    for (int c = 0; c < Cin; ++c)
        s += xb[(size_t)c * Nin] * wb[(size_t)c * Cout * 2];
    cat[((size_t)b * CatC + o) * Nout + t] = s;
}

// ---------------------------------------------------------------------------
// Copy skip tensor into channels [C, 2C) of the concat buffer.
// ---------------------------------------------------------------------------
__global__ __launch_bounds__(256)
void copy_skip_kernel(const float* __restrict__ s, float* __restrict__ cat,
                      int C, int N, int CatC, long long total)
{
    long long i = (long long)blockIdx.x * 256 + threadIdx.x;
    if (i >= total) return;                    // total = B*C*N
    int n = (int)(i % N);
    long long r = i / N;
    int c = (int)(r % C);
    int b = (int)(r / C);
    cat[((size_t)b * CatC + C + c) * N + n] = s[i];
}

// ---------------------------------------------------------------------------
// Final conv1d(k=1, Cin=16, Cout=1) + sigmoid.
// ---------------------------------------------------------------------------
__global__ __launch_bounds__(256)
void outconv_sigmoid_kernel(const float* __restrict__ x,
                            const float* __restrict__ w,
                            const float* __restrict__ bias,
                            float* __restrict__ y, int N, long long total)
{
    long long i = (long long)blockIdx.x * 256 + threadIdx.x;
    if (i >= total) return;                    // total = B*N
    int n = (int)(i % N);
    int b = (int)(i / N);
    float s = bias[0];
    #pragma unroll
    for (int c = 0; c < 16; ++c)
        s += w[c] * x[((size_t)b * 16 + c) * N + n];
    y[i] = 1.0f / (1.0f + __expf(-s));
}

// ---------------------------------------------------------------------------
// Host side
// ---------------------------------------------------------------------------
struct BlockP {
    const float *w1, *b1, *g1, *be1, *m1, *v1;
    const float *w2, *b2, *g2, *be2, *m2, *v2;
};
struct DecP { const float *uw, *ub; BlockP blk; };

static const int B = 64;

static void launch_block(hipStream_t s, const float* in, float* tmp, float* out,
                         const BlockP& p, int Cin, int Cout, int N)
{
    dim3 g(N / 256, Cout / 16, B);
    conv_bn_relu_wmma<<<g, 128, 0, s>>>(in, p.w1, p.b1, p.g1, p.be1, p.m1, p.v1,
                                        tmp, Cin, Cout, N);
    conv_bn_relu_wmma<<<g, 128, 0, s>>>(tmp, p.w2, p.b2, p.g2, p.be2, p.m2, p.v2,
                                        out, Cout, Cout, N);
}

static inline long long cdiv256(long long t) { return (t + 255) / 256; }

extern "C" void kernel_launch(void* const* d_in, const int* in_sizes, int n_in,
                              void* d_out, int out_size, void* d_ws, size_t ws_size,
                              hipStream_t stream)
{
    (void)in_sizes; (void)n_in; (void)out_size; (void)ws_size;

    // ---- parse inputs in setup_inputs() dict order ----
    int cur = 0;
    auto nf = [&]() { return (const float*)d_in[cur++]; };
    const float* x = nf();

    BlockP e1, e2, e3, e4, bo;
    DecP   d1, d2, d3, d4;
    auto rb = [&](BlockP& p) {
        p.w1 = nf(); p.b1 = nf(); p.g1 = nf(); p.be1 = nf(); p.m1 = nf(); p.v1 = nf();
        p.w2 = nf(); p.b2 = nf(); p.g2 = nf(); p.be2 = nf(); p.m2 = nf(); p.v2 = nf();
    };
    auto rd = [&](DecP& p) { p.uw = nf(); p.ub = nf(); rb(p.blk); };
    rb(e1); rb(e2); rb(e3); rb(e4); rb(bo);
    rd(d1); rd(d2); rd(d3); rd(d4);
    const float* ow = nf();
    const float* ob = nf();

    // ---- workspace layout (floats) ----
    float* ws = (float*)d_ws;
    size_t off = 0;
    float* X0  = ws + off; off += (size_t)B * 1 * 16384;          // ssa(x)
    float* T1  = ws + off; off += (size_t)16777216;               // block tmp
    float* S1  = ws + off; off += (size_t)16777216;               // skips
    float* S2  = ws + off; off += (size_t)16777216;
    float* S3  = ws + off; off += (size_t)16777216;
    float* S4  = ws + off; off += (size_t)16777216;
    float* P   = ws + off; off += (size_t)8388608;                // pooled
    float* CAT = ws + off; off += (size_t)33554432;               // concat
    float* D   = ws + off; off += (size_t)16777216;               // dec/bottom out

    // ---- encoder ----
    ssa_kernel<<<B, 256, 0, stream>>>(x, X0, 1, 16384);
    launch_block(stream, X0, T1, S1, e1, 1, 16, 16384);

    { long long t = (long long)B * 16 * 8192;
      pool_kernel<<<cdiv256(t), 256, 0, stream>>>(S1, P, 8192, t); }
    ssa_kernel<<<B, 256, 0, stream>>>(P, P, 16, 8192);
    launch_block(stream, P, T1, S2, e2, 16, 32, 8192);

    { long long t = (long long)B * 32 * 4096;
      pool_kernel<<<cdiv256(t), 256, 0, stream>>>(S2, P, 4096, t); }
    ssa_kernel<<<B, 256, 0, stream>>>(P, P, 32, 4096);
    launch_block(stream, P, T1, S3, e3, 32, 64, 4096);

    { long long t = (long long)B * 64 * 2048;
      pool_kernel<<<cdiv256(t), 256, 0, stream>>>(S3, P, 2048, t); }
    ssa_kernel<<<B, 256, 0, stream>>>(P, P, 64, 2048);
    launch_block(stream, P, T1, S4, e4, 64, 128, 2048);

    { long long t = (long long)B * 128 * 1024;
      pool_kernel<<<cdiv256(t), 256, 0, stream>>>(S4, P, 1024, t); }   // no ssa
    launch_block(stream, P, T1, D, bo, 128, 256, 1024);

    // ---- decoder: up -> concat(skip) -> block ----
    struct Step { DecP* p; float* skip; int cin; int cout; int nin; };
    Step steps[4] = {
        { &d1, S4, 256, 128, 1024 },
        { &d2, S3, 128,  64, 2048 },
        { &d3, S2,  64,  32, 4096 },
        { &d4, S1,  32,  16, 8192 },
    };
    for (int i = 0; i < 4; ++i) {
        Step& s = steps[i];
        int nout = 2 * s.nin;
        int catc = 2 * s.cout;
        long long tu = (long long)B * s.cout * nout;
        upconv_kernel<<<cdiv256(tu), 256, 0, stream>>>(
            D, s.p->uw, s.p->ub, CAT, s.cin, s.cout, s.nin, catc, tu);
        long long tc = (long long)B * s.cout * nout;
        copy_skip_kernel<<<cdiv256(tc), 256, 0, stream>>>(
            s.skip, CAT, s.cout, nout, catc, tc);
        launch_block(stream, CAT, T1, D, s.p->blk, catc, s.cout, nout);
    }

    // ---- 1x1 conv + sigmoid -> d_out ----
    { long long t = (long long)B * 16384;
      outconv_sigmoid_kernel<<<cdiv256(t), 256, 0, stream>>>(
          D, ow, ob, (float*)d_out, 16384, t); }
}
